// SwinDiffusionEncoder_20590073217469
// MI455X (gfx1250) — matrix operand order repaired and checked by reading, not compiled, not run on hardware
//
#include <hip/hip_runtime.h>
#include <math.h>

typedef _Float16 f16;
typedef __attribute__((ext_vector_type(4)))  f16   h4;
typedef __attribute__((ext_vector_type(8)))  f16   v8h;
typedef __attribute__((ext_vector_type(16))) f16   v16h;
typedef __attribute__((ext_vector_type(8)))  float v8f;

enum : int { ACT_NONE = 0, ACT_GELU = 1, ACT_SILU = 2 };

// ---------------------------------------------------------------------------
// WMMA fragment loader: p[0..7] and p[off2..off2+7] (halves) -> v16h
// ---------------------------------------------------------------------------
__device__ inline v16h ld_frag(const f16* p, int off2) {
  union { v16h v; v8h h[2]; } u;
  u.h[0] = *(const v8h*)(p);
  u.h[1] = *(const v8h*)(p + off2);
  return u.v;
}

// CDNA5 async global->LDS copy (16B per lane), tracked by ASYNCcnt.
__device__ inline void async_ld_b128(unsigned lds_off, const void* gaddr) {
  asm volatile("global_load_async_to_lds_b128 %0, %1, off"
               :: "v"(lds_off), "v"(gaddr)
               : "memory");
}
__device__ inline void wait_async0() {
  asm volatile("s_wait_asynccnt 0x0" ::: "memory");
}

// ---------------------------------------------------------------------------
// Tiled f16 WMMA GEMM: C = act(A(MxK) * B(KxN) + bias)
// Block 256 threads = 8 waves in 4x2 grid, each wave 32x32 output.
// Double-buffered LDS; A tile filled with GLOBAL_LOAD_ASYNC_TO_LDS_B128 so
// the k+1 tile fill overlaps the k tile's WMMAs.
// ---------------------------------------------------------------------------
constexpr int GBM = 128, GBN = 64, GBK = 32;
constexpr int GLDA = GBK + 8;   // 40 halves = 80B rows (16B aligned)
constexpr int GLDB = GBK + 8;   // B stored transposed [n][k]

template <int ACT, bool OUTH>
__global__ __launch_bounds__(256) void wmma_gemm(
    const f16* __restrict__ A, const f16* __restrict__ B,
    const float* __restrict__ bias, void* __restrict__ Cout,
    int M, int N, int K) {
  __shared__ __align__(16) f16 As[2][GBM * GLDA];
  __shared__ __align__(16) f16 Bs[2][GBN * GLDB];

  const int tid = threadIdx.x;
  const int lane = tid & 31, wid = tid >> 5;
  const int wm = wid >> 1, wn = wid & 1;
  const int bm = blockIdx.x * GBM, bn = blockIdx.y * GBN;
  const int mloc = lane & 15;
  const int khalf = lane >> 4;
  const int nk = (K + GBK - 1) / GBK;

  auto load_tiles = [&](int kt, int buf) {
    const int k0 = kt * GBK;
    f16* as = &As[buf][0];
    f16* bs = &Bs[buf][0];
    if (bm + GBM <= M && k0 + GBK <= K) {
      // fast path: async copy, 16B per lane, 2 chunks per thread
      for (int v = 0; v < 2; ++v) {
        int linear = tid + v * 256;      // 512 oct-chunks of 8 halves
        int r = linear >> 2;             // 4 chunks per 32-half row
        int c = (linear & 3) * 8;
        unsigned lds = (unsigned)(size_t)(as + r * GLDA + c);
        const f16* g = A + (size_t)(bm + r) * K + k0 + c;
        async_ld_b128(lds, g);
      }
    } else {
      for (int v = 0; v < 4; ++v) {
        int linear = tid + v * 256;      // quad index
        int r = linear >> 3;
        int c = (linear & 7) * 4;
        h4 val = {};
        int gr = bm + r;
        if (gr < M) {
          int gc = k0 + c;
          if (gc + 3 < K) {
            val = *(const h4*)(A + (size_t)gr * K + gc);
          } else {
            for (int i = 0; i < 4; ++i)
              val[i] = (gc + i < K) ? A[(size_t)gr * K + gc + i] : (f16)0;
          }
        }
        *(h4*)(as + r * GLDA + c) = val;
      }
    }
    // B tile, stored transposed Bs[n][k]
    for (int v = 0; v < 8; ++v) {
      int linear = tid + v * 256;        // 0..2047
      int kk = linear >> 6;
      int nn = linear & 63;
      f16 val = (f16)0;
      if (bn + nn < N && k0 + kk < K) val = B[(size_t)(k0 + kk) * N + bn + nn];
      bs[nn * GLDB + kk] = val;
      if (v == 0 && k0 + GBK + kk < K && bn + nn < N)
        __builtin_prefetch(B + (size_t)(k0 + GBK + kk) * N + bn + nn, 0, 1);
    }
  };

  v8f acc[2][2] = {};

  load_tiles(0, 0);
  wait_async0();
  __syncthreads();

  for (int kt = 0; kt < nk; ++kt) {
    int cur = kt & 1;
    if (kt + 1 < nk) load_tiles(kt + 1, cur ^ 1);

    const f16* as = &As[cur][0];
    const f16* bs = &Bs[cur][0];
    v16h afr[2], bfr[2];
    for (int f = 0; f < 2; ++f) {
      afr[f] = ld_frag(as + (wm * 32 + f * 16 + mloc) * GLDA + khalf * 8, 16);
      bfr[f] = ld_frag(bs + (wn * 32 + f * 16 + mloc) * GLDB + khalf * 8, 16);
    }
    for (int fm = 0; fm < 2; ++fm)
      for (int fn = 0; fn < 2; ++fn)
        acc[fm][fn] = __builtin_amdgcn_wmma_f32_16x16x32_f16(
            false, afr[fm], false, bfr[fn], (short)0, acc[fm][fn], false, false);

    wait_async0();
    __syncthreads();
  }

  const int rowoff = khalf * 8;
  const int ncol = mloc;
  for (int fm = 0; fm < 2; ++fm)
    for (int fn = 0; fn < 2; ++fn)
      for (int r = 0; r < 8; ++r) {
        int gr = bm + wm * 32 + fm * 16 + rowoff + r;
        int gc = bn + wn * 32 + fn * 16 + ncol;
        if (gr < M && gc < N) {
          float v = acc[fm][fn][r];
          if (bias) v += bias[gc];
          if (ACT == ACT_GELU) v = 0.5f * v * (1.0f + erff(v * 0.70710678f));
          else if (ACT == ACT_SILU) v = v / (1.0f + expf(-v));
          if (OUTH) ((f16*)Cout)[(size_t)gr * N + gc] = (f16)v;
          else      ((float*)Cout)[(size_t)gr * N + gc] = v;
        }
      }
}

// ---------------------------------------------------------------------------
// SwinV2 window attention (cosine sim, rel-bias, shift mask), WMMA for N>=16.
// grid = (numWindowsTotal, nh), block = 256.
// qkv rows: win*N + i, cols: (s*nh+h)*d + e   (3C total)
// out rows: win*N + i, cols: h*d + e          (C total)
// ---------------------------------------------------------------------------
__device__ inline int win_label(int t, int wy, int wx, int wsz, int Hn, int Wn,
                                int shift) {
  int r = wy * wsz + t / wsz;
  int c = wx * wsz + t % wsz;
  int rr = (r < Hn - wsz) ? 0 : ((r < Hn - shift) ? 1 : 2);
  int cc = (c < Wn - wsz) ? 0 : ((c < Wn - shift) ? 1 : 2);
  return rr * 3 + cc;
}

__global__ __launch_bounds__(256) void k_attn(
    const f16* __restrict__ qkv, const float* __restrict__ bias,
    const float* __restrict__ lsc, f16* __restrict__ out,
    int N, int d, int nh, int Hn, int Wn, int wsz, int shift, int nWw, int nW) {
  __shared__ __align__(16) f16 Qs[64 * 72];
  __shared__ __align__(16) f16 Ks[64 * 72];
  __shared__ __align__(16) f16 Vst[64 * 72];   // transposed: [e][token]
  __shared__ __align__(16) f16 Ps[64 * 72];
  __shared__ float Ss[64 * 65];
  __shared__ float qn[64], kn[64];

  const int tid = threadIdx.x;
  const int lane = tid & 31, wid = tid >> 5;
  const int mloc = lane & 15, khalf = lane >> 4;
  const int win = blockIdx.x, h = blockIdx.y;
  const int wIdx = win % nW;
  const int wy = wIdx / nWw, wx = wIdx % nWw;
  const int C3 = 3 * nh * d;
  const long rowbase = (long)win * N;

  for (int i = tid; i < 64 * 72; i += 256) {
    Qs[i] = (f16)0; Ks[i] = (f16)0; Vst[i] = (f16)0; Ps[i] = (f16)0;
  }
  if (tid < 64) { qn[tid] = 0.f; kn[tid] = 0.f; }
  __syncthreads();

  for (int idx = tid; idx < N * d; idx += 256) {
    int i = idx / d, e = idx % d;
    const f16* base = qkv + (rowbase + i) * C3;
    f16 qv = base[h * d + e];
    f16 kv = base[nh * d + h * d + e];
    f16 vv = base[2 * nh * d + h * d + e];
    Qs[i * 72 + e] = qv;
    Ks[i * 72 + e] = kv;
    Vst[e * 72 + i] = vv;
    float qf = (float)qv, kf = (float)kv;
    atomicAdd(&qn[i], qf * qf);
    atomicAdd(&kn[i], kf * kf);
  }
  __syncthreads();
  for (int idx = tid; idx < N * d; idx += 256) {
    int i = idx / d, e = idx % d;
    float qi = 1.f / fmaxf(sqrtf(qn[i]), 1e-12f);
    float ki = 1.f / fmaxf(sqrtf(kn[i]), 1e-12f);
    Qs[i * 72 + e] = (f16)((float)Qs[i * 72 + e] * qi);
    Ks[i * 72 + e] = (f16)((float)Ks[i * 72 + e] * ki);
  }
  __syncthreads();

  const float scale = expf(fminf(lsc[h], 4.6051702f));  // log(100)

  if (N >= 16) {
    const int nt = N / 16;          // 1 or 4
    const int nks = d / 32;         // 1 or 2
    for (int t = wid; t < nt * nt; t += 8) {
      int ti = t / nt, tj = t % nt;
      v8f acc = {};
      for (int ks = 0; ks < nks; ++ks) {
        v16h a = ld_frag(&Qs[(ti * 16 + mloc) * 72 + ks * 32 + khalf * 8], 16);
        v16h b = ld_frag(&Ks[(tj * 16 + mloc) * 72 + ks * 32 + khalf * 8], 16);
        acc = __builtin_amdgcn_wmma_f32_16x16x32_f16(false, a, false, b,
                                                     (short)0, acc, false, false);
      }
      for (int r = 0; r < 8; ++r) {
        int gi = ti * 16 + khalf * 8 + r;
        int gj = tj * 16 + mloc;
        float v;
        if (gi < N && gj < N) {
          float m = 0.f;
          if (shift > 0 &&
              win_label(gi, wy, wx, wsz, Hn, Wn, shift) !=
              win_label(gj, wy, wx, wsz, Hn, Wn, shift))
            m = -100.f;
          v = acc[r] * scale + bias[(h * N + gi) * N + gj] + m;
        } else {
          v = -1e30f;
        }
        Ss[gi * 65 + gj] = v;
      }
    }
    __syncthreads();

    if (tid < N) {
      int r = tid;
      float mx = -1e30f;
      for (int j = 0; j < N; ++j) mx = fmaxf(mx, Ss[r * 65 + j]);
      float sum = 0.f;
      for (int j = 0; j < N; ++j) { float e = expf(Ss[r * 65 + j] - mx); Ss[r * 65 + j] = e; sum += e; }
      float inv = 1.f / sum;
      for (int j = 0; j < N; ++j) Ps[r * 72 + j] = (f16)(Ss[r * 65 + j] * inv);
    }
    __syncthreads();

    const int ct = d / 16;
    const int nks2 = (N > 32) ? 2 : 1;
    for (int t = wid; t < nt * ct; t += 8) {
      int ti = t / ct, tj = t % ct;
      v8f acc = {};
      for (int ks = 0; ks < nks2; ++ks) {
        v16h a = ld_frag(&Ps[(ti * 16 + mloc) * 72 + ks * 32 + khalf * 8], 16);
        v16h b = ld_frag(&Vst[(tj * 16 + mloc) * 72 + ks * 32 + khalf * 8], 16);
        acc = __builtin_amdgcn_wmma_f32_16x16x32_f16(false, a, false, b,
                                                     (short)0, acc, false, false);
      }
      for (int r = 0; r < 8; ++r) {
        int gi = ti * 16 + khalf * 8 + r;
        int gc = tj * 16 + mloc;
        if (gi < N && gc < d)
          out[(rowbase + gi) * (nh * d) + h * d + gc] = (f16)acc[r];
      }
    }
  } else {
    // tiny windows (N==4): scalar path
    if (tid < N) {
      int i = tid;
      float srow[16];
      float mx = -1e30f;
      for (int j = 0; j < N; ++j) {
        float s = 0.f;
        for (int e = 0; e < d; ++e) s += (float)Qs[i * 72 + e] * (float)Ks[j * 72 + e];
        float m = 0.f;
        if (shift > 0 &&
            win_label(i, wy, wx, wsz, Hn, Wn, shift) !=
            win_label(j, wy, wx, wsz, Hn, Wn, shift))
          m = -100.f;
        s = s * scale + bias[(h * N + i) * N + j] + m;
        srow[j] = s; mx = fmaxf(mx, s);
      }
      float sum = 0.f;
      for (int j = 0; j < N; ++j) { srow[j] = expf(srow[j] - mx); sum += srow[j]; }
      float inv = 1.f / sum;
      for (int e = 0; e < d; ++e) {
        float o = 0.f;
        for (int j = 0; j < N; ++j) o += srow[j] * (float)Vst[e * 72 + j];
        out[(rowbase + i) * (nh * d) + h * d + e] = (f16)(o * inv);
      }
    }
  }
}

// ---------------------------------------------------------------------------
// Support kernels
// ---------------------------------------------------------------------------
__global__ void k_f32_to_f16(const float* __restrict__ in, f16* __restrict__ out,
                             long long n) {
  long long i = (long long)blockIdx.x * 256 + threadIdx.x;
  if (i < n) out[i] = (f16)in[i];
}

// in (O,K) -> out (K,O)
__global__ void k_transpose_to_f16(const float* __restrict__ in,
                                   f16* __restrict__ out, int O, int K) {
  long long i = (long long)blockIdx.x * 256 + threadIdx.x;
  if (i >= (long long)O * K) return;
  int o = (int)(i % O); int k = (int)(i / O);
  out[i] = (f16)in[(size_t)o * K + k];
}

__global__ void k_concat_qkv_bias(const float* qb, const float* vb, float* out,
                                  int C) {
  int i = blockIdx.x * 256 + threadIdx.x;
  if (i >= 3 * C) return;
  out[i] = (i < C) ? qb[i] : ((i < 2 * C) ? 0.f : vb[i - 2 * C]);
}

__global__ void k_window_gather(const float* __restrict__ x, f16* __restrict__ out,
                                int B, int H, int W, int C, int wsz, int shift) {
  long long total = (long long)B * H * W * C;
  long long i = (long long)blockIdx.x * 256 + threadIdx.x;
  if (i >= total) return;
  int c = (int)(i % C); long long r = i / C;
  int L = H * W; int b = (int)(r / L); int rem = (int)(r % L);
  int nWw = W / wsz, N = wsz * wsz;
  int win = rem / N, t = rem % N;
  int wy = win / nWw, wx = win % nWw;
  int hh = wy * wsz + t / wsz, ww = wx * wsz + t % wsz;
  int sh = (hh + shift) % H, sw = (ww + shift) % W;
  out[i] = (f16)x[((size_t)(b * H + sh) * W + sw) * C + c];
}

__global__ void k_window_reverse(const float* __restrict__ pr,
                                 float* __restrict__ y, int B, int H, int W,
                                 int C, int wsz, int shift) {
  long long total = (long long)B * H * W * C;
  long long i = (long long)blockIdx.x * 256 + threadIdx.x;
  if (i >= total) return;
  int c = (int)(i % C); long long r = i / C;
  int L = H * W; int b = (int)(r / L); int rem = (int)(r % L);
  int hh = rem / W, ww = rem % W;
  int sh = (hh - shift + H) % H, sw = (ww - shift + W) % W;
  int nWw = W / wsz, N = wsz * wsz;
  int wy = sh / wsz, wx = sw / wsz;
  int t = (sh % wsz) * wsz + (sw % wsz);
  long long row = (long long)b * L + (wy * nWw + wx) * N + t;
  y[i] = pr[row * C + c];
}

// im2col for PxP stride-P patch conv: rows = B*(HO*HO), K = Cin*P*P
__global__ void k_patch_im2col(const float* __restrict__ x, f16* __restrict__ out,
                               int B, int Cin, int IMGsz, int P) {
  int HO = IMGsz / P; int Kd = Cin * P * P;
  long long total = (long long)B * HO * HO * Kd;
  long long i = (long long)blockIdx.x * 256 + threadIdx.x;
  if (i >= total) return;
  int k = (int)(i % Kd); long long r = i / Kd;
  int b = (int)(r / (HO * HO)); int pix = (int)(r % (HO * HO));
  int oy = pix / HO, ox = pix % HO;
  int cin = k / (P * P); int kh = (k % (P * P)) / P; int kw = k % P;
  out[i] = (f16)x[(((size_t)b * Cin + cin) * IMGsz + oy * P + kh) * IMGsz + ox * P + kw];
}

// im2col for 2x2 stride-2 merge conv on (B,H,W,C): K = C*4, k = c*4 + kh*2 + kw
__global__ void k_merge_conv_im2col(const float* __restrict__ x,
                                    f16* __restrict__ out, int B, int H, int W,
                                    int C) {
  int HO = H / 2, WO = W / 2; int Kd = 4 * C;
  long long total = (long long)B * HO * WO * Kd;
  long long i = (long long)blockIdx.x * 256 + threadIdx.x;
  if (i >= total) return;
  int k = (int)(i % Kd); long long r = i / Kd;
  int b = (int)(r / (HO * WO)); int pix = (int)(r % (HO * WO));
  int oy = pix / WO, ox = pix % WO;
  int c = k / 4, kh = (k % 4) / 2, kw = k % 2;
  out[i] = (f16)x[((size_t)(b * H + oy * 2 + kh) * W + ox * 2 + kw) * C + c];
}

// patch-merge linear gather: k = part*C + c, parts (0,0),(1,0),(0,1),(1,1)
__global__ void k_merge_lin_gather(const float* __restrict__ x,
                                   f16* __restrict__ out, int B, int H, int W,
                                   int C) {
  int HO = H / 2, WO = W / 2; int Kd = 4 * C;
  long long total = (long long)B * HO * WO * Kd;
  long long i = (long long)blockIdx.x * 256 + threadIdx.x;
  if (i >= total) return;
  int k = (int)(i % Kd); long long r = i / Kd;
  int b = (int)(r / (HO * WO)); int pix = (int)(r % (HO * WO));
  int oy = pix / WO, ox = pix % WO;
  int part = k / C, c = k % C;
  const int dhs[4] = {0, 1, 0, 1};
  const int dws[4] = {0, 0, 1, 1};
  out[i] = (f16)x[((size_t)(b * H + oy * 2 + dhs[part]) * W + ox * 2 + dws[part]) * C + c];
}

__global__ __launch_bounds__(256) void k_gn_stats(const float* __restrict__ y,
                                                  float* __restrict__ stats,
                                                  int B, int L, int C, int G) {
  int bg = blockIdx.x; int g = bg % G; int b = bg / G;
  int Cg = C / G; long long cnt = (long long)L * Cg;
  float s1 = 0.f, s2 = 0.f;
  for (long long idx = threadIdx.x; idx < cnt; idx += 256) {
    int l = (int)(idx / Cg); int c = g * Cg + (int)(idx % Cg);
    float v = y[((size_t)b * L + l) * C + c];
    s1 += v; s2 += v * v;
  }
  __shared__ float r1[256], r2[256];
  r1[threadIdx.x] = s1; r2[threadIdx.x] = s2;
  __syncthreads();
  for (int s = 128; s > 0; s >>= 1) {
    if ((int)threadIdx.x < s) { r1[threadIdx.x] += r1[threadIdx.x + s]; r2[threadIdx.x] += r2[threadIdx.x + s]; }
    __syncthreads();
  }
  if (threadIdx.x == 0) {
    float m = r1[0] / (float)cnt;
    float var = r2[0] / (float)cnt - m * m;
    stats[2 * bg] = m;
    stats[2 * bg + 1] = rsqrtf(fmaxf(var, 0.f) + 1e-5f);
  }
}

__global__ void k_gn_apply(const float* __restrict__ y, const float* __restrict__ stats,
                           const float* __restrict__ gamma, const float* __restrict__ beta,
                           const float* __restrict__ resid, const float* __restrict__ extra,
                           float* __restrict__ of, f16* __restrict__ oh,
                           int B, int L, int C, int G, int act) {
  long long total = (long long)B * L * C;
  long long i = (long long)blockIdx.x * 256 + threadIdx.x;
  if (i >= total) return;
  int c = (int)(i % C); long long bl = i / C; int b = (int)(bl / L);
  int Cg = C / G; int g = c / Cg;
  float m = stats[2 * (b * G + g)], inv = stats[2 * (b * G + g) + 1];
  float t = (y[i] - m) * inv * gamma[c] + beta[c];
  if (extra) t += extra[i % ((long long)L * C)];
  if (resid) t += resid[i];
  if (act == ACT_SILU) t = t / (1.f + expf(-t));
  of[i] = t;
  if (oh) oh[i] = (f16)t;
}

// continuous rel-pos bias MLP (2 -> 512 relu -> nh), then sigmoid gather
__global__ void k_cpb_hidden(const float* __restrict__ w1, const float* __restrict__ b1,
                             float* __restrict__ hid, int wsz) {
  int R = (2 * wsz - 1) * (2 * wsz - 1);
  long long total = (long long)R * 512;
  long long i = (long long)blockIdx.x * 256 + threadIdx.x;
  if (i >= total) return;
  int j = (int)(i % 512); int r = (int)(i / 512);
  float ra = (float)(r / (2 * wsz - 1) - (wsz - 1));
  float rb = (float)(r % (2 * wsz - 1) - (wsz - 1));
  float sc = 8.f / fmaxf((float)(wsz - 1), 1.f);
  float ta = ra * sc, tb = rb * sc;
  float fa = copysignf(log2f(fabsf(ta) + 1.f) / log2f(8.f), ta);
  float fb = copysignf(log2f(fabsf(tb) + 1.f) / log2f(8.f), tb);
  float v = fa * w1[j] + fb * w1[512 + j] + b1[j];
  hid[i] = fmaxf(v, 0.f);
}

__global__ void k_cpb_bt(const float* __restrict__ hid, const float* __restrict__ w2,
                         float* __restrict__ bt, int R, int nh) {
  long long i = (long long)blockIdx.x * 128 + threadIdx.x;
  if (i >= (long long)R * nh) return;
  int h = (int)(i % nh); int r = (int)(i / nh);
  float s = 0.f;
  for (int j = 0; j < 512; ++j) s += hid[(size_t)r * 512 + j] * w2[(size_t)j * nh + h];
  bt[i] = s;
}

__global__ void k_cpb_gather(const float* __restrict__ bt, float* __restrict__ bias,
                             int wsz, int nh) {
  int N = wsz * wsz;
  long long total = (long long)nh * N * N;
  long long i = (long long)blockIdx.x * 256 + threadIdx.x;
  if (i >= total) return;
  int h = (int)(i / (N * N)); int rem = (int)(i % (N * N));
  int ti = rem / N, tj = rem % N;
  int yi = ti / wsz, xi = ti % wsz, yj = tj / wsz, xj = tj % wsz;
  int ridx = (yi - yj + wsz - 1) * (2 * wsz - 1) + (xi - xj + wsz - 1);
  float v = bt[(size_t)ridx * nh + h];
  bias[i] = 16.f / (1.f + expf(-v));
}

__global__ void k_pool_mean_concat(const float* __restrict__ y, const float* __restrict__ pos,
                                   float* __restrict__ ycf, f16* __restrict__ ych,
                                   int B, int L, int C) {
  long long total = (long long)B * (L + 1) * C;
  long long i = (long long)blockIdx.x * 256 + threadIdx.x;
  if (i >= total) return;
  int c = (int)(i % C); long long r = i / C;
  int b = (int)(r / (L + 1)); int t = (int)(r % (L + 1));
  float v;
  if (t == 0) {
    v = 0.f;
    for (int l = 0; l < L; ++l) v += y[((size_t)b * L + l) * C + c];
    v *= (1.f / (float)L);
  } else {
    v = y[((size_t)b * L + (t - 1)) * C + c];
  }
  v += pos[(size_t)t * C + c];
  ycf[i] = v; ych[i] = (f16)v;
}

__global__ void k_pool_attn(const float* __restrict__ Q, const float* __restrict__ K,
                            const float* __restrict__ V, f16* __restrict__ O,
                            int B, int C, int heads) {
  int bh = blockIdx.x; int b = bh / heads; int h = bh % heads;
  int d = C / heads;
  int e = threadIdx.x;   // blockDim == d
  __shared__ float sc[32];
  if (e < 17) {
    const float* q = Q + (size_t)(b * 17) * C + h * d;
    const float* k = K + (size_t)(b * 17 + e) * C + h * d;
    float s = 0.f;
    for (int t = 0; t < d; ++t) s += q[t] * k[t];
    sc[e] = s * rsqrtf((float)d);
  }
  __syncthreads();
  float mx = -1e30f;
  for (int j = 0; j < 17; ++j) mx = fmaxf(mx, sc[j]);
  float a[17]; float sum = 0.f;
  for (int j = 0; j < 17; ++j) { a[j] = expf(sc[j] - mx); sum += a[j]; }
  float o = 0.f;
  for (int j = 0; j < 17; ++j) o += a[j] * V[(size_t)(b * 17 + j) * C + h * d + e];
  O[(size_t)b * C + h * d + e] = (f16)(o / sum);
}

// ---------------------------------------------------------------------------
// Host orchestration
// ---------------------------------------------------------------------------
static inline long long cdivll(long long a, long long b) { return (a + b - 1) / b; }

static void gemm(hipStream_t st, const f16* A, const f16* B, const float* bias,
                 void* C, int M, int N, int K, int act, bool outHalf) {
  dim3 g((unsigned)((M + GBM - 1) / GBM), (unsigned)((N + GBN - 1) / GBN));
  dim3 b(256);
  if (outHalf) {
    if (act == ACT_GELU)
      wmma_gemm<ACT_GELU, true><<<g, b, 0, st>>>(A, B, bias, C, M, N, K);
    else
      wmma_gemm<ACT_NONE, true><<<g, b, 0, st>>>(A, B, bias, C, M, N, K);
  } else {
    wmma_gemm<ACT_NONE, false><<<g, b, 0, st>>>(A, B, bias, C, M, N, K);
  }
}

struct BlockP {
  const float *qkv_w, *q_bias, *v_bias, *logit_scale, *cpb_w1, *cpb_b1, *cpb_w2,
      *proj_w, *proj_b, *norm1_g, *norm1_b, *fc1_w, *fc1_b, *fc2_w, *fc2_b,
      *norm2_g, *norm2_b;
};

struct Bufs {
  float *XF, *XF2, *YF, *PR, *BQ, *STATS, *CPBH, *BT, *BIAS, *YCF, *QF, *KF, *VF;
  f16 *XH, *XW, *QKV, *AO, *H16, *W16, *YCH, *OH;
};

static void run_block(hipStream_t st, Bufs& w, const BlockP& p, int B, int H,
                      int Wd, int C, int nh, int wsz, int shift) {
  int L = H * Wd, R = B * L, N = wsz * wsz;
  int nWw = Wd / wsz, nW = (H / wsz) * nWw, d = C / nh;
  long long RC = (long long)R * C;

  // window partition (+shift roll), f32 -> f16
  k_window_gather<<<(unsigned)cdivll(RC, 256), 256, 0, st>>>(w.XF, w.XW, B, H, Wd, C, wsz, shift);
  // qkv projection
  k_concat_qkv_bias<<<(unsigned)cdivll(3 * C, 256), 256, 0, st>>>(p.q_bias, p.v_bias, w.BQ, C);
  k_f32_to_f16<<<(unsigned)cdivll((long long)C * 3 * C, 256), 256, 0, st>>>(p.qkv_w, w.W16, (long long)C * 3 * C);
  gemm(st, w.XW, w.W16, w.BQ, w.QKV, R, 3 * C, C, ACT_NONE, true);
  // continuous rel-pos bias
  int Rt = (2 * wsz - 1) * (2 * wsz - 1);
  k_cpb_hidden<<<(unsigned)cdivll((long long)Rt * 512, 256), 256, 0, st>>>(p.cpb_w1, p.cpb_b1, w.CPBH, wsz);
  k_cpb_bt<<<(unsigned)cdivll((long long)Rt * nh, 128), 128, 0, st>>>(w.CPBH, p.cpb_w2, w.BT, Rt, nh);
  k_cpb_gather<<<(unsigned)cdivll((long long)nh * N * N, 256), 256, 0, st>>>(w.BT, w.BIAS, wsz, nh);
  // attention
  dim3 ag((unsigned)(B * nW), (unsigned)nh);
  k_attn<<<ag, 256, 0, st>>>(w.QKV, w.BIAS, p.logit_scale, w.AO, N, d, nh, H, Wd, wsz, shift, nWw, nW);
  // output projection
  k_f32_to_f16<<<(unsigned)cdivll((long long)C * C, 256), 256, 0, st>>>(p.proj_w, w.W16, (long long)C * C);
  gemm(st, w.AO, w.W16, p.proj_b, w.PR, R, C, C, ACT_NONE, false);
  // reverse windows (+roll back)
  k_window_reverse<<<(unsigned)cdivll(RC, 256), 256, 0, st>>>(w.PR, w.YF, B, H, Wd, C, wsz, shift);
  // x = shortcut + GN(attn_out)
  k_gn_stats<<<(unsigned)(B * nh), 256, 0, st>>>(w.YF, w.STATS, B, L, C, nh);
  k_gn_apply<<<(unsigned)cdivll(RC, 256), 256, 0, st>>>(w.YF, w.STATS, p.norm1_g, p.norm1_b, w.XF, nullptr, w.XF2, w.XH, B, L, C, nh, ACT_NONE);
  { float* t = w.XF; w.XF = w.XF2; w.XF2 = t; }
  // MLP
  k_f32_to_f16<<<(unsigned)cdivll((long long)C * 4 * C, 256), 256, 0, st>>>(p.fc1_w, w.W16, (long long)C * 4 * C);
  gemm(st, w.XH, w.W16, p.fc1_b, w.H16, R, 4 * C, C, ACT_GELU, true);
  k_f32_to_f16<<<(unsigned)cdivll((long long)4 * C * C, 256), 256, 0, st>>>(p.fc2_w, w.W16, (long long)4 * C * C);
  gemm(st, w.H16, w.W16, p.fc2_b, w.YF, R, C, 4 * C, ACT_NONE, false);
  k_gn_stats<<<(unsigned)(B * nh), 256, 0, st>>>(w.YF, w.STATS, B, L, C, nh);
  k_gn_apply<<<(unsigned)cdivll(RC, 256), 256, 0, st>>>(w.YF, w.STATS, p.norm2_g, p.norm2_b, w.XF, nullptr, w.XF2, w.XH, B, L, C, nh, ACT_NONE);
  { float* t = w.XF; w.XF = w.XF2; w.XF2 = t; }
}

extern "C" void kernel_launch(void* const* d_in, const int* in_sizes, int n_in,
                              void* d_out, int out_size, void* d_ws, size_t ws_size,
                              hipStream_t stream) {
  (void)in_sizes; (void)n_in; (void)ws_size;
  const int Bn = 8;
  const int HEADS[4] = {3, 6, 12, 24};
  const int WSZ[4] = {8, 4, 4, 2};

  int ci = 0;
  auto F = [&]() -> const float* { return (const float*)d_in[ci++]; };
  auto readBlock = [&]() {
    BlockP p;
    p.qkv_w = F(); p.q_bias = F(); p.v_bias = F(); p.logit_scale = F();
    p.cpb_w1 = F(); p.cpb_b1 = F(); p.cpb_w2 = F();
    p.proj_w = F(); p.proj_b = F(); p.norm1_g = F(); p.norm1_b = F();
    p.fc1_w = F(); p.fc1_b = F(); p.fc2_w = F(); p.fc2_b = F();
    p.norm2_g = F(); p.norm2_b = F();
    return p;
  };

  const float* x = F();
  const float* patch_w = F(); const float* patch_b = F();
  const float* pn_g = F(); const float* pn_b = F();
  const float* abs_pos = F();
  BlockP initb[2]; initb[0] = readBlock(); initb[1] = readBlock();
  BlockP l1[4][2], l2[4][2];
  const float *mc_w = 0, *mc_b = 0;
  const float *mr[4] = {0, 0, 0, 0}, *mg[4], *mb2[4];
  for (int s = 0; s < 4; ++s) {
    l1[s][0] = readBlock(); l1[s][1] = readBlock();
    if (s == 0) { mc_w = F(); mc_b = F(); mg[0] = F(); mb2[0] = F(); }
    else { mr[s] = F(); mg[s] = F(); mb2[s] = F(); }
    l2[s][0] = readBlock(); l2[s][1] = readBlock();
  }
  BlockP midb[2]; midb[0] = readBlock(); midb[1] = readBlock();
  const float* pl_ng = F(); const float* pl_nb = F(); const float* pl_pos = F();
  const float* q_w = F(); const float* q_b = F();
  const float* k_w = F(); const float* k_b = F();
  const float* v_w = F(); const float* v_b = F();
  const float* c_w = F(); const float* c_b = F();

  // workspace carve
  char* wp = (char*)d_ws;
  auto carve = [&](size_t bytes) -> void* {
    void* p = wp; wp += (bytes + 255) & ~(size_t)255; return p;
  };
  const size_t MAXE = 3145728;    // 8*4096*96
  Bufs w;
  w.XF  = (float*)carve(MAXE * 4);
  w.XF2 = (float*)carve(MAXE * 4);
  w.YF  = (float*)carve(MAXE * 4);
  w.PR  = (float*)carve(MAXE * 4);
  w.XH  = (f16*)carve(MAXE * 2);
  w.XW  = (f16*)carve(MAXE * 2);
  w.AO  = (f16*)carve(MAXE * 2);
  w.QKV = (f16*)carve((size_t)9437184 * 2);
  w.H16 = (f16*)carve((size_t)12582912 * 2);
  w.W16 = (f16*)carve((size_t)9437184 * 2);
  w.BQ  = (float*)carve(4608 * 4);
  w.STATS = (float*)carve(1024 * 4);
  w.CPBH  = (float*)carve(115200 * 4);
  w.BT    = (float*)carve(8192 * 4);
  w.BIAS  = (float*)carve(16384 * 4);
  w.YCF = (float*)carve(208896 * 4);
  w.YCH = (f16*)carve(208896 * 2);
  w.QF  = (float*)carve(208896 * 4);
  w.KF  = (float*)carve(208896 * 4);
  w.VF  = (float*)carve(208896 * 4);
  w.OH  = (f16*)carve(12288 * 2);

  // ---- patch embedding: 4x4/4 conv as im2col GEMM (K=48, N=96) ----
  int H = 64, Wd = 64, dim = 96;
  long long L0 = (long long)H * Wd;
  k_patch_im2col<<<(unsigned)cdivll((long long)Bn * L0 * 48, 256), 256, 0, stream>>>(x, w.XW, Bn, 3, 256, 4);
  k_transpose_to_f16<<<(unsigned)cdivll(96 * 48, 256), 256, 0, stream>>>(patch_w, w.W16, 96, 48);
  gemm(stream, w.XW, w.W16, patch_b, w.YF, (int)(Bn * L0), 96, 48, ACT_NONE, false);
  k_gn_stats<<<(unsigned)(Bn * 3), 256, 0, stream>>>(w.YF, w.STATS, Bn, (int)L0, 96, 3);
  k_gn_apply<<<(unsigned)cdivll((long long)Bn * L0 * 96, 256), 256, 0, stream>>>(
      w.YF, w.STATS, pn_g, pn_b, nullptr, abs_pos, w.XF, w.XH, Bn, (int)L0, 96, 3, ACT_NONE);

  // ---- init blocks ----
  for (int i = 0; i < 2; ++i)
    run_block(stream, w, initb[i], Bn, H, Wd, dim, HEADS[0], WSZ[0], (i % 2) ? WSZ[0] / 2 : 0);

  // ---- 4 stages ----
  for (int s = 0; s < 4; ++s) {
    for (int i = 0; i < 2; ++i)
      run_block(stream, w, l1[s][i], Bn, H, Wd, dim, HEADS[s], WSZ[s], (i % 2) ? WSZ[s] / 2 : 0);
    int L = H * Wd;
    long long rows = (long long)Bn * (L / 4);
    if (s == 0) {
      k_merge_conv_im2col<<<(unsigned)cdivll(rows * 4 * dim, 256), 256, 0, stream>>>(w.XF, w.XW, Bn, H, Wd, dim);
      k_transpose_to_f16<<<(unsigned)cdivll((long long)2 * dim * 4 * dim, 256), 256, 0, stream>>>(mc_w, w.W16, 2 * dim, 4 * dim);
      gemm(stream, w.XW, w.W16, mc_b, w.YF, (int)rows, 2 * dim, 4 * dim, ACT_NONE, false);
    } else {
      k_merge_lin_gather<<<(unsigned)cdivll(rows * 4 * dim, 256), 256, 0, stream>>>(w.XF, w.XW, Bn, H, Wd, dim);
      k_f32_to_f16<<<(unsigned)cdivll((long long)4 * dim * 2 * dim, 256), 256, 0, stream>>>(mr[s], w.W16, (long long)4 * dim * 2 * dim);
      gemm(stream, w.XW, w.W16, nullptr, w.YF, (int)rows, 2 * dim, 4 * dim, ACT_NONE, false);
    }
    k_gn_stats<<<(unsigned)(Bn * HEADS[s]), 256, 0, stream>>>(w.YF, w.STATS, Bn, L / 4, 2 * dim, HEADS[s]);
    k_gn_apply<<<(unsigned)cdivll(rows * 2 * dim, 256), 256, 0, stream>>>(
        w.YF, w.STATS, mg[s], mb2[s], nullptr, nullptr, w.XF, w.XH, Bn, L / 4, 2 * dim, HEADS[s], ACT_NONE);
    dim *= 2; H /= 2; Wd /= 2;
    for (int i = 0; i < 2; ++i)
      run_block(stream, w, l2[s][i], Bn, H, Wd, dim, HEADS[s], WSZ[s], (i % 2) ? WSZ[s] / 2 : 0);
  }

  // ---- mid blocks (dim=1536, H=W=4) ----
  for (int i = 0; i < 2; ++i)
    run_block(stream, w, midb[i], Bn, H, Wd, dim, HEADS[3], WSZ[3], (i % 2) ? WSZ[3] / 2 : 0);

  // ---- attention pooling head ----
  int Lf = H * Wd;  // 16
  k_gn_stats<<<(unsigned)(Bn * HEADS[3]), 256, 0, stream>>>(w.XF, w.STATS, Bn, Lf, dim, HEADS[3]);
  k_gn_apply<<<(unsigned)cdivll((long long)Bn * Lf * dim, 256), 256, 0, stream>>>(
      w.XF, w.STATS, pl_ng, pl_nb, nullptr, nullptr, w.YF, nullptr, Bn, Lf, dim, HEADS[3], ACT_SILU);
  k_pool_mean_concat<<<(unsigned)cdivll((long long)Bn * (Lf + 1) * dim, 256), 256, 0, stream>>>(
      w.YF, pl_pos, w.YCF, w.YCH, Bn, Lf, dim);
  int Rp = Bn * (Lf + 1);
  k_f32_to_f16<<<(unsigned)cdivll((long long)dim * dim, 256), 256, 0, stream>>>(q_w, w.W16, (long long)dim * dim);
  gemm(stream, w.YCH, w.W16, q_b, w.QF, Rp, dim, dim, ACT_NONE, false);
  k_f32_to_f16<<<(unsigned)cdivll((long long)dim * dim, 256), 256, 0, stream>>>(k_w, w.W16, (long long)dim * dim);
  gemm(stream, w.YCH, w.W16, k_b, w.KF, Rp, dim, dim, ACT_NONE, false);
  k_f32_to_f16<<<(unsigned)cdivll((long long)dim * dim, 256), 256, 0, stream>>>(v_w, w.W16, (long long)dim * dim);
  gemm(stream, w.YCH, w.W16, v_b, w.VF, Rp, dim, dim, ACT_NONE, false);
  k_pool_attn<<<(unsigned)(Bn * 8), 192, 0, stream>>>(w.QF, w.KF, w.VF, w.OH, Bn, dim, 8);
  k_f32_to_f16<<<(unsigned)cdivll((long long)dim * 1024, 256), 256, 0, stream>>>(c_w, w.W16, (long long)dim * 1024);
  gemm(stream, w.OH, w.W16, c_b, d_out, Bn, 1024, dim, ACT_NONE, false);
  (void)out_size;
}
